// CausalSelfAttention_40630390620994
// MI455X (gfx1250) — compile-verified, matrix operand order checked by reading, not run on hardware
//
#include <hip/hip_runtime.h>
#include <hip/hip_bf16.h>

#define BB 2
#define SS 4096
#define DD 512
#define HH 8
#define HDIM 64
#define BSROWS (BB*SS)          // 8192

typedef __attribute__((ext_vector_type(16))) __bf16 v16bf;
typedef __attribute__((ext_vector_type(8)))  __bf16 v8bf;
typedef __attribute__((ext_vector_type(8)))  float  v8f;

static __device__ __forceinline__ v8f wmma_bf16(v16bf a, v16bf b, v8f c) {
    // D = A(16x32 bf16) x B(32x16 bf16) + C(16x16 f32)
    return __builtin_amdgcn_wmma_f32_16x16x32_bf16(false, a, false, b, (short)0, c, false, false);
}

static __device__ __forceinline__ v8f zero8() {
    v8f z;
#pragma unroll
    for (int i = 0; i < 8; ++i) z[i] = 0.0f;
    return z;
}

// ---------------------------------------------------------------------------
// Kernel 0a: weights W[in][out] f32 -> Wt[mat][out][in] bf16 via LDS transpose
// ---------------------------------------------------------------------------
__global__ __launch_bounds__(256) void wconv_kernel(
        const float* __restrict__ Wq, const float* __restrict__ Wk,
        const float* __restrict__ Wv, const float* __restrict__ Wo,
        __bf16* __restrict__ wt) {
    __shared__ __bf16 tile[32][34];
    const int bid = blockIdx.x;
    const int mat = bid >> 8;
    const int t2  = bid & 255;
    const int i0 = (t2 >> 4) * 32;
    const int j0 = (t2 & 15) * 32;
    const float* W = (mat == 0) ? Wq : (mat == 1) ? Wk : (mat == 2) ? Wv : Wo;
    const int c = threadIdx.x & 31;
    const int r = threadIdx.x >> 5;
#pragma unroll
    for (int k = 0; k < 4; ++k) {
        const int row = r + k * 8;
        tile[row][c] = (__bf16)W[(size_t)(i0 + row) * DD + j0 + c];
    }
    __syncthreads();
    __bf16* Wt = wt + (size_t)mat * (DD * DD);
#pragma unroll
    for (int k = 0; k < 4; ++k) {
        const int row = r + k * 8;
        Wt[(size_t)(j0 + row) * DD + i0 + c] = tile[c][row];
    }
}

// ---------------------------------------------------------------------------
// Kernel 0b: x f32 -> bf16, packed 16B stores
// ---------------------------------------------------------------------------
__global__ __launch_bounds__(256) void xconv_kernel(
        const float* __restrict__ x, __bf16* __restrict__ xb) {
    const int idx = blockIdx.x * 256 + threadIdx.x;
    const size_t base = (size_t)idx * 8;
    float4 f0 = *(const float4*)(x + base);
    float4 f1 = *(const float4*)(x + base + 4);
    v8bf o;
    o[0] = (__bf16)f0.x; o[1] = (__bf16)f0.y; o[2] = (__bf16)f0.z; o[3] = (__bf16)f0.w;
    o[4] = (__bf16)f1.x; o[5] = (__bf16)f1.y; o[6] = (__bf16)f1.z; o[7] = (__bf16)f1.w;
    *(v8bf*)(xb + base) = o;
}

// ===========================================================================
// Shared GEMM inner machinery: 32x64 tile per wave, DD-length K dim,
// ping-pong double buffering (no register rotation copies).
// ===========================================================================
#define LOAD_A(dst, off) \
    do { (dst)[0] = *(const v16bf*)(aptr0 + (off)); \
         (dst)[1] = *(const v16bf*)(aptr1 + (off)); } while (0)
#define LOAD_B(dst, off) \
    do { _Pragma("unroll") \
         for (int t_ = 0; t_ < 4; ++t_) \
             (dst)[t_] = *(const v16bf*)(bptr + (size_t)t_ * 16 * DD + (off)); } while (0)
#define WMMA8(abuf, bbuf) \
    do { _Pragma("unroll") \
         for (int t_ = 0; t_ < 4; ++t_) { \
             acc[0][t_] = wmma_bf16((abuf)[0], (bbuf)[t_], acc[0][t_]); \
             acc[1][t_] = wmma_bf16((abuf)[1], (bbuf)[t_], acc[1][t_]); } } while (0)

// ---------------------------------------------------------------------------
// Kernel 1: QKV projections.  grid = (BSROWS/256, 24)
// ---------------------------------------------------------------------------
__global__ __launch_bounds__(256) void qkv_kernel(
        const __bf16* __restrict__ xb, const __bf16* __restrict__ wt,
        const float* __restrict__ bq, const float* __restrict__ bk,
        const float* __restrict__ bv,
        __bf16* __restrict__ Q, __bf16* __restrict__ K, __bf16* __restrict__ Vt) {
    const int lane = threadIdx.x & 31;
    const int wave = threadIdx.x >> 5;
    const int ll = lane & 15, lh = lane >> 4;
    const int proj = blockIdx.y >> 3;              // 0=q 1=k 2=v
    const int n0   = (blockIdx.y & 7) * 64;
    const int row0 = blockIdx.x * 256 + wave * 32;
    const __bf16* W = wt + (size_t)proj * (DD * DD);
    const float* bias = (proj == 0) ? bq : (proj == 1) ? bk : bv;

    const __bf16* aptr0 = xb + (size_t)(row0 + ll) * DD + lh * 16;
    const __bf16* aptr1 = aptr0 + (size_t)16 * DD;
    const __bf16* bptr  = W + (size_t)(n0 + ll) * DD + lh * 16;

    v8f acc[2][4];
#pragma unroll
    for (int i = 0; i < 2; ++i)
#pragma unroll
        for (int t = 0; t < 4; ++t) acc[i][t] = zero8();

    v16bf a0[2], b0[4], a1[2], b1[4];
    LOAD_A(a0, 0); LOAD_B(b0, 0);
    for (int k0 = 0; k0 < DD - 64; k0 += 64) {
        LOAD_A(a1, k0 + 32); LOAD_B(b1, k0 + 32);
        WMMA8(a0, b0);
        LOAD_A(a0, k0 + 64); LOAD_B(b0, k0 + 64);
        WMMA8(a1, b1);
    }
    LOAD_A(a1, DD - 32); LOAD_B(b1, DD - 32);
    WMMA8(a0, b0);
    WMMA8(a1, b1);

#pragma unroll
    for (int t = 0; t < 4; ++t) {
        const int col = n0 + t * 16 + ll;
        const float bcol = bias[col];
        const int h = col >> 6, hd = col & 63;
#pragma unroll
        for (int i = 0; i < 2; ++i) {
            const int rbase = row0 + i * 16 + lh * 8;
            const int b_ = rbase >> 12;
            const int s0 = rbase & 4095;
            if (proj == 2) {
                v8bf pk;
#pragma unroll
                for (int r = 0; r < 8; ++r) pk[r] = (__bf16)(acc[i][t][r] + bcol);
                *(v8bf*)(Vt + ((size_t)(b_ * HH + h) * HDIM + hd) * SS + s0) = pk;
            } else {
                const float qs = (proj == 0) ? 0.125f : 1.0f;
                __bf16* dst = (proj == 0) ? Q : K;
#pragma unroll
                for (int r = 0; r < 8; ++r) {
                    dst[((size_t)(b_ * HH + h) * SS + (s0 + r)) * HDIM + hd] =
                        (__bf16)((acc[i][t][r] + bcol) * qs);
                }
            }
        }
    }
}

// ---------------------------------------------------------------------------
// Kernel 2: per-wave causal flash attention, ping-pong K prefetch.
//   grid = (SS/128, B*H), block 256 (8 waves), 16 query rows per wave.
// ---------------------------------------------------------------------------
__global__ __launch_bounds__(256) void attn_kernel(
        const __bf16* __restrict__ Q, const __bf16* __restrict__ K,
        const __bf16* __restrict__ Vt, __bf16* __restrict__ O) {
    __shared__ __align__(64) __bf16 plds[8][16][32];

    const int lane = threadIdx.x & 31;
    const int wave = threadIdx.x >> 5;
    const int ll = lane & 15, lh = lane >> 4;
    const int bh = blockIdx.y;
    const int b_ = bh >> 3, h = bh & 7;
    const int q0 = blockIdx.x * 128 + wave * 16;

    const __bf16* Qh = Q  + (size_t)bh * SS * HDIM;
    const __bf16* Kh = K  + (size_t)bh * SS * HDIM;
    const __bf16* Vh = Vt + (size_t)bh * HDIM * SS;

    v16bf qf[2];
#pragma unroll
    for (int kc = 0; kc < 2; ++kc)
        qf[kc] = *(const v16bf*)(Qh + (size_t)(q0 + ll) * HDIM + kc * 32 + lh * 16);

    v8f acc[4];
#pragma unroll
    for (int t = 0; t < 4; ++t) acc[t] = zero8();
    float m[8], l[8];
#pragma unroll
    for (int r = 0; r < 8; ++r) { m[r] = -3.0e38f; l[r] = 0.0f; }

    const int jtmax = (q0 + 15) >> 5;

    auto loadK = [&](int jt, v16bf (&dst)[4]) {
        const int kv0 = jt * 32;
#pragma unroll
        for (int t = 0; t < 2; ++t)
#pragma unroll
            for (int kc = 0; kc < 2; ++kc)
                dst[t * 2 + kc] = *(const v16bf*)(Kh + (size_t)(kv0 + t * 16 + ll) * HDIM + kc * 32 + lh * 16);
    };

    // one kv block: S=Q*K^T from kcur, prefetch K[jtpre] into knext,
    // online softmax, P via LDS, O += P*V
    auto step = [&](int jt, v16bf (&kcur)[4], v16bf (&knext)[4], int jtpre) {
        const int kv0 = jt * 32;
        // V fragments issued early, consumed after softmax
        v16bf vf[4];
#pragma unroll
        for (int nt = 0; nt < 4; ++nt)
            vf[nt] = *(const v16bf*)(Vh + (size_t)(nt * 16 + ll) * SS + kv0 + lh * 16);

        v8f st[2];
#pragma unroll
        for (int t = 0; t < 2; ++t) {
            v8f s = zero8();
            s = wmma_bf16(qf[0], kcur[t * 2 + 0], s);
            s = wmma_bf16(qf[1], kcur[t * 2 + 1], s);
            st[t] = s;
        }

        loadK(jtpre, knext);

#pragma unroll
        for (int r = 0; r < 8; ++r) {
            const int row = q0 + lh * 8 + r;
            float s0 = st[0][r];
            float s1 = st[1][r];
            if (kv0 + ll > row)      s0 = -1.0e30f;
            if (kv0 + 16 + ll > row) s1 = -1.0e30f;
            float mx = fmaxf(s0, s1);
            mx = fmaxf(mx, __shfl_xor(mx, 1, 32));
            mx = fmaxf(mx, __shfl_xor(mx, 2, 32));
            mx = fmaxf(mx, __shfl_xor(mx, 4, 32));
            mx = fmaxf(mx, __shfl_xor(mx, 8, 32));
            const float mn = fmaxf(m[r], mx);
            const float alpha = __expf(m[r] - mn);
            m[r] = mn;
            const float p0 = __expf(s0 - mn);
            const float p1 = __expf(s1 - mn);
            float rs = p0 + p1;
            rs += __shfl_xor(rs, 1, 32);
            rs += __shfl_xor(rs, 2, 32);
            rs += __shfl_xor(rs, 4, 32);
            rs += __shfl_xor(rs, 8, 32);
            l[r] = l[r] * alpha + rs;
#pragma unroll
            for (int t = 0; t < 4; ++t) acc[t][r] *= alpha;
            plds[wave][lh * 8 + r][ll]      = (__bf16)p0;
            plds[wave][lh * 8 + r][16 + ll] = (__bf16)p1;
        }
        v16bf pf = *(const v16bf*)(&plds[wave][ll][lh * 16]);
#pragma unroll
        for (int nt = 0; nt < 4; ++nt)
            acc[nt] = wmma_bf16(pf, vf[nt], acc[nt]);
    };

    v16bf kfA[4], kfB[4];
    loadK(0, kfA);
    int jt = 0;
    for (; jt + 1 <= jtmax; jt += 2) {
        step(jt, kfA, kfB, jt + 1);                                 // uses A, fills B
        step(jt + 1, kfB, kfA, (jt + 2 <= jtmax) ? jt + 2 : jtmax); // uses B, fills A
    }
    if (jt <= jtmax)
        step(jt, kfA, kfB, jt);   // tail (prefetch target unused)

    // normalize and store O (bf16, [b][s][D])
#pragma unroll
    for (int r = 0; r < 8; ++r) {
        const float inv = 1.0f / l[r];
        const int s_ = q0 + lh * 8 + r;
#pragma unroll
        for (int nt = 0; nt < 4; ++nt) {
            O[((size_t)(b_ * SS + s_)) * DD + h * HDIM + nt * 16 + ll] =
                (__bf16)(acc[nt][r] * inv);
        }
    }
}

// ---------------------------------------------------------------------------
// Kernel 3: out = O @ Wo + bo (f32).  grid = (BSROWS/256, 8)
// ---------------------------------------------------------------------------
__global__ __launch_bounds__(256) void oproj_kernel(
        const __bf16* __restrict__ O, const __bf16* __restrict__ wt,
        const float* __restrict__ bo, float* __restrict__ out) {
    const int lane = threadIdx.x & 31;
    const int wave = threadIdx.x >> 5;
    const int ll = lane & 15, lh = lane >> 4;
    const int row0 = blockIdx.x * 256 + wave * 32;
    const int n0 = blockIdx.y * 64;
    const __bf16* W = wt + (size_t)3 * (DD * DD);   // Wo^T

    const __bf16* aptr0 = O + (size_t)(row0 + ll) * DD + lh * 16;
    const __bf16* aptr1 = aptr0 + (size_t)16 * DD;
    const __bf16* bptr  = W + (size_t)(n0 + ll) * DD + lh * 16;

    v8f acc[2][4];
#pragma unroll
    for (int i = 0; i < 2; ++i)
#pragma unroll
        for (int t = 0; t < 4; ++t) acc[i][t] = zero8();

    v16bf a0[2], b0[4], a1[2], b1[4];
    LOAD_A(a0, 0); LOAD_B(b0, 0);
    for (int k0 = 0; k0 < DD - 64; k0 += 64) {
        LOAD_A(a1, k0 + 32); LOAD_B(b1, k0 + 32);
        WMMA8(a0, b0);
        LOAD_A(a0, k0 + 64); LOAD_B(b0, k0 + 64);
        WMMA8(a1, b1);
    }
    LOAD_A(a1, DD - 32); LOAD_B(b1, DD - 32);
    WMMA8(a0, b0);
    WMMA8(a1, b1);

#pragma unroll
    for (int t = 0; t < 4; ++t) {
        const int col = n0 + t * 16 + ll;
        const float bc = bo[col];
#pragma unroll
        for (int i = 0; i < 2; ++i) {
            const int rbase = row0 + i * 16 + lh * 8;
#pragma unroll
            for (int r = 0; r < 8; ++r)
                out[(size_t)(rbase + r) * DD + col] = acc[i][t][r] + bc;
        }
    }
}

// ---------------------------------------------------------------------------
extern "C" void kernel_launch(void* const* d_in, const int* in_sizes, int n_in,
                              void* d_out, int out_size, void* d_ws, size_t ws_size,
                              hipStream_t stream) {
    const float* x  = (const float*)d_in[0];
    const float* Wq = (const float*)d_in[1];
    const float* bq = (const float*)d_in[2];
    const float* Wk = (const float*)d_in[3];
    const float* bk = (const float*)d_in[4];
    const float* Wv = (const float*)d_in[5];
    const float* bv = (const float*)d_in[6];
    const float* Wo = (const float*)d_in[7];
    const float* bo = (const float*)d_in[8];
    float* out = (float*)d_out;

    char* ws = (char*)d_ws;
    __bf16* wt = (__bf16*)(ws);                              //  2 MB: Wt bf16 [mat][out][in]
    __bf16* xb = (__bf16*)(ws + ( 2ull << 20));              //  8 MB: x bf16
    __bf16* Qw = (__bf16*)(ws + (10ull << 20));              //  8 MB: Q [b][h][s][hd] (pre-scaled)
    __bf16* Kw = (__bf16*)(ws + (18ull << 20));              //  8 MB: K [b][h][s][hd]
    __bf16* Vw = (__bf16*)(ws + (26ull << 20));              //  8 MB: V^T [b][h][hd][s]
    __bf16* Ow = (__bf16*)(ws + (34ull << 20));              //  8 MB: attn out [b][s][D]

    wconv_kernel<<<dim3(4 * 16 * 16), dim3(256), 0, stream>>>(Wq, Wk, Wv, Wo, wt);
    xconv_kernel<<<dim3(BSROWS * DD / 8 / 256), dim3(256), 0, stream>>>(x, xb);
    qkv_kernel<<<dim3(BSROWS / 256, 24), dim3(256), 0, stream>>>(xb, wt, bq, bk, bv, Qw, Kw, Vw);
    attn_kernel<<<dim3(SS / 128, BB * HH), dim3(256), 0, stream>>>(Qw, Kw, Vw, Ow);
    oproj_kernel<<<dim3(BSROWS / 256, 8), dim3(256), 0, stream>>>(Ow, wt, bo, out);
}
